// DFineMultiheadAttention_8839042695545
// MI455X (gfx1250) — matrix operand order repaired and checked by reading, not compiled, not run on hardware
//
#include <hip/hip_runtime.h>
#include <hip/hip_bf16.h>
#include <cstdint>

typedef __attribute__((ext_vector_type(16))) _Float16 v16h;
typedef __attribute__((ext_vector_type(8)))  _Float16 v8h;
typedef __attribute__((ext_vector_type(8)))  float    v8f;

#define B_  4
#define S_  2048
#define D_  256
#define H_  8
#define HD_ 32
#define SCALING 0.17677669529663687f   // 32^-0.5

// ---------------------------------------------------------------------------
// WMMA helpers (gfx1250: V_WMMA_F32_16X16X32_F16, wave32)
// A fragment (16x32 f16): lane m = lane&15, elem i -> K = kk + g*16 + i,
//                         g = lane>>4
// B fragment (32x16 f16): lane n = lane&15, elem i -> K = kk + g*16 + i
// C/D (16x16 f32, 8 VGPR): reg r -> row = r + 8*g, col = lane&15
// ---------------------------------------------------------------------------
__device__ __forceinline__ v16h load_ab(const _Float16* p0, const _Float16* p1) {
  v8h x = *(const v8h*)p0;
  v8h y = *(const v8h*)p1;
  v16h r;
#pragma unroll
  for (int i = 0; i < 8; ++i) { r[i] = x[i]; r[i + 8] = y[i]; }
  return r;
}

__device__ __forceinline__ v8f wmma16(v16h a, v16h b, v8f c) {
  return __builtin_amdgcn_wmma_f32_16x16x32_f16(false, a, false, b, (short)0, c,
                                                false, false);
}

// 32x32 output tile of A[M,256] * Bt[N,256]^T, K = 256 (8 k-steps, 4 WMMA each)
__device__ __forceinline__ void gemm_tile_32x32(const _Float16* __restrict__ A,
                                                const _Float16* __restrict__ Bt,
                                                int row0, int col0, int lane,
                                                v8f& c00, v8f& c01,
                                                v8f& c10, v8f& c11) {
  int col = lane & 15;
  int g   = lane >> 4;
  const _Float16* ap0 = A  + (size_t)(row0 + col) * 256;
  const _Float16* ap1 = A  + (size_t)(row0 + 16 + col) * 256;
  const _Float16* bp0 = Bt + (size_t)(col0 + col) * 256;
  const _Float16* bp1 = Bt + (size_t)(col0 + 16 + col) * 256;
  v8f z = {};
  c00 = z; c01 = z; c10 = z; c11 = z;
#pragma unroll
  for (int kk = 0; kk < 256; kk += 32) {
    v16h a0 = load_ab(ap0 + kk + g * 16, ap0 + kk + g * 16 + 8);
    v16h a1 = load_ab(ap1 + kk + g * 16, ap1 + kk + g * 16 + 8);
    v16h b0 = *(const v16h*)(bp0 + kk + g * 16);
    v16h b1 = *(const v16h*)(bp1 + kk + g * 16);
    c00 = wmma16(a0, b0, c00);
    c01 = wmma16(a0, b1, c01);
    c10 = wmma16(a1, b0, c10);
    c11 = wmma16(a1, b1, c11);
  }
}

// ---------------------------------------------------------------------------
// Kernel 1: hp = hidden + pos -> f16 ; hidden -> f16
// ---------------------------------------------------------------------------
__global__ void pack_act_kernel(const float* __restrict__ hid,
                                const float* __restrict__ pos,
                                _Float16* __restrict__ hpf,
                                _Float16* __restrict__ hf, int n) {
  for (int i = blockIdx.x * blockDim.x + threadIdx.x; i < n;
       i += gridDim.x * blockDim.x) {
    float h = hid[i];
    hpf[i] = (_Float16)(h + pos[i]);
    hf[i]  = (_Float16)h;
  }
}

// ---------------------------------------------------------------------------
// Kernel 2: transpose + f16-convert the four 256x256 weight matrices
// ---------------------------------------------------------------------------
__global__ void pack_w_kernel(const float* __restrict__ Wq,
                              const float* __restrict__ Wk,
                              const float* __restrict__ Wv,
                              const float* __restrict__ Wo,
                              _Float16* __restrict__ Wqt,
                              _Float16* __restrict__ Wkt,
                              _Float16* __restrict__ Wvt,
                              _Float16* __restrict__ Wot) {
  int i = blockIdx.x * blockDim.x + threadIdx.x;   // 65536 elems: i = k*256 + n
  int k = i >> 8, n = i & 255;
  int t = n * 256 + k;
  Wqt[t] = (_Float16)Wq[i];
  Wkt[t] = (_Float16)Wk[i];
  Wvt[t] = (_Float16)Wv[i];
  Wot[t] = (_Float16)Wo[i];
}

// ---------------------------------------------------------------------------
// Kernel 3: QKV projections. One wave = one 32x32 tile of [8192 x 256].
// which 0: Q = (hp*Wq + bq)*scale  -> [B,H,S,HD]
// which 1: K =  hp*Wk + bk         -> [B,H,S,HD]
// which 2: V =  h *Wv + bv         -> [B,H,HD,S]  (transposed for PV GEMM)
// ---------------------------------------------------------------------------
__global__ void __launch_bounds__(128)
qkv_proj_kernel(const _Float16* __restrict__ hpf, const _Float16* __restrict__ hf,
                const _Float16* __restrict__ Wqt, const _Float16* __restrict__ Wkt,
                const _Float16* __restrict__ Wvt,
                const float* __restrict__ bq, const float* __restrict__ bk,
                const float* __restrict__ bv,
                _Float16* __restrict__ Qf, _Float16* __restrict__ Kf,
                _Float16* __restrict__ Vt) {
  int lane  = threadIdx.x & 31;
  int wf    = blockIdx.x * 4 + (threadIdx.x >> 5);
  int which = wf >> 11;            // 2048 (32x32)-tiles per matrix
  int rem   = wf & 2047;
  int mt    = rem >> 3;            // 256 row-tiles of 32
  int nt    = rem & 7;             // 8 col-tiles of 32
  int g = lane >> 4, col = lane & 15;

  const _Float16* A    = (which == 2) ? hf : hpf;
  const _Float16* Wt   = (which == 0) ? Wqt : (which == 1) ? Wkt : Wvt;
  const float*    bias = (which == 0) ? bq  : (which == 1) ? bk  : bv;

  v8f c[2][2];
  gemm_tile_32x32(A, Wt, mt * 32, nt * 32, lane, c[0][0], c[0][1], c[1][0], c[1][1]);

  int hh = nt;                     // n>>5 == nt since n-tile is 32-aligned
#pragma unroll
  for (int sn = 0; sn < 2; ++sn) {
    int   n    = nt * 32 + sn * 16 + col;
    float bval = bias[n];
    int   e    = sn * 16 + col;    // n & 31
#pragma unroll
    for (int sm = 0; sm < 2; ++sm) {
#pragma unroll
      for (int r = 0; r < 8; ++r) {
        int   gm = mt * 32 + sm * 16 + r + 8 * g;
        int   b  = gm >> 11;       // /S_
        int   s  = gm & 2047;
        float v  = c[sm][sn][r] + bval;
        if (which == 0)
          Qf[(((size_t)b * H_ + hh) * S_ + s) * HD_ + e] = (_Float16)(v * SCALING);
        else if (which == 1)
          Kf[(((size_t)b * H_ + hh) * S_ + s) * HD_ + e] = (_Float16)v;
        else
          Vt[(((size_t)b * H_ + hh) * HD_ + e) * S_ + s] = (_Float16)v;
      }
    }
  }
}

// ---------------------------------------------------------------------------
// Kernel 4: flash attention. One wave = one 16-row q-tile of one (b,h).
// k-chunks of 32: 2 WMMA scores, online softmax (max via shuffles, sum via
// a P x ones WMMA), 2 WMMA ctx. Mask stream prefetched 2 chunks ahead.
// ---------------------------------------------------------------------------
__global__ void __launch_bounds__(128)
flash_attn_kernel(const _Float16* __restrict__ Qf, const _Float16* __restrict__ Kf,
                  const _Float16* __restrict__ Vt, const float* __restrict__ mask,
                  _Float16* __restrict__ ctx) {
  __shared__ __align__(16) _Float16 sP[4][16 * 32];   // 1KB per wave
  int lane  = threadIdx.x & 31;
  int w     = threadIdx.x >> 5;
  int idx   = blockIdx.x * 4 + w;                     // 0..4095
  int qtile = idx & 127;                              // S/16
  int h     = (idx >> 7) & 7;
  int b     = idx >> 10;
  int col = lane & 15, g = lane >> 4;

  const _Float16* Qb = Qf + (size_t)(b * H_ + h) * S_ * HD_;
  const _Float16* Kb = Kf + (size_t)(b * H_ + h) * S_ * HD_;
  const _Float16* Vb = Vt + (size_t)(b * H_ + h) * HD_ * S_;
  int q0 = qtile * 16;

  // Q A-fragment: row = q0 + (lane&15)
  const _Float16* qp = Qb + (size_t)(q0 + col) * HD_;
  v16h qa = load_ab(qp + g * 16, qp + g * 16 + 8);

  // all-ones B fragment: rowsum(P) = P x ones lands in C layout matching lrow
  v16h ones;
#pragma unroll
  for (int i = 0; i < 16; ++i) ones[i] = (_Float16)1.0f;

  v8f acc0 = {}, acc1 = {};
  v8f zf = {};
  float mrow[8], lrow[8], arow[8];
#pragma unroll
  for (int r = 0; r < 8; ++r) { mrow[r] = -1.0e30f; lrow[r] = 0.0f; }

  _Float16* sp = sP[w];
  const float* mbase = mask + ((size_t)b * S_ + q0) * S_;

  for (int kk = 0; kk < S_; kk += 32) {
    // prefetch mask two chunks ahead (wraps are harmless speculative lines)
    if (kk + 64 < S_)
      __builtin_prefetch(mbase + (size_t)col * S_ + kk + 64 + g * 16, 0, 0);

    // K^T B-fragments: col n = key (lane&15), elem i -> hd = g*16 + i
    v16h kb0 = *(const v16h*)(Kb + (size_t)(kk + col) * HD_ + g * 16);
    v16h kb1 = *(const v16h*)(Kb + (size_t)(kk + 16 + col) * HD_ + g * 16);
    v8f s0 = wmma16(qa, kb0, zf);    // scores for keys kk..kk+15
    v8f s1 = wmma16(qa, kb1, zf);    // scores for keys kk+16..kk+31

#pragma unroll
    for (int r = 0; r < 8; ++r) {
      int rq = r + 8 * g;            // C-layout row within tile
      const float* mp = mbase + (size_t)rq * S_ + kk;
      float x0 = s0[r] + mp[col];
      float x1 = s1[r] + mp[16 + col];
      float mx = fmaxf(x0, x1);      // row-max across 16 lanes of the group
      mx = fmaxf(mx, __shfl_xor(mx, 1, 32));
      mx = fmaxf(mx, __shfl_xor(mx, 2, 32));
      mx = fmaxf(mx, __shfl_xor(mx, 4, 32));
      mx = fmaxf(mx, __shfl_xor(mx, 8, 32));
      float mnew  = fmaxf(mrow[r], mx);
      float alpha = __expf(mrow[r] - mnew);
      float p0 = __expf(x0 - mnew);
      float p1 = __expf(x1 - mnew);
      mrow[r] = mnew;
      arow[r] = alpha;
      acc0[r] *= alpha;
      acc1[r] *= alpha;
      sp[rq * 32 + col]      = (_Float16)p0;   // stage P (C-layout -> LDS)
      sp[rq * 32 + 16 + col] = (_Float16)p1;
    }
    __syncthreads();                 // uniform across block (same trip count)
    // Reload P as an A-fragment (row = lane&15, K = chunk-local key)
    const _Float16* pp = sp + col * 32;
    v16h pa = load_ab(pp + g * 16, pp + g * 16 + 8);
    __syncthreads();                 // protect sP before next iteration writes

    // row sums via matrix pipe instead of 32 shuffles
    v8f rs = wmma16(pa, ones, zf);
#pragma unroll
    for (int r = 0; r < 8; ++r) lrow[r] = lrow[r] * arow[r] + rs[r];

    // V^T B-fragments: col n = hd half (lane&15), elem i -> key = g*16 + i
    v16h vb0 = *(const v16h*)(Vb + (size_t)col * S_ + kk + g * 16);
    v16h vb1 = *(const v16h*)(Vb + (size_t)(16 + col) * S_ + kk + g * 16);
    acc0 = wmma16(pa, vb0, acc0);
    acc1 = wmma16(pa, vb1, acc1);
  }

#pragma unroll
  for (int r = 0; r < 8; ++r) {
    float inv = 1.0f / lrow[r];
    int   rq  = r + 8 * g;
    size_t ro = ((size_t)b * S_ + (q0 + rq)) * (size_t)D_ + (size_t)h * HD_;
    ctx[ro + col]      = (_Float16)(acc0[r] * inv);
    ctx[ro + 16 + col] = (_Float16)(acc1[r] * inv);
  }
}

// ---------------------------------------------------------------------------
// Kernel 5: out = ctx * Wo + bo   (fp32 output), 32x32 tile per wave
// ---------------------------------------------------------------------------
__global__ void __launch_bounds__(128)
out_proj_kernel(const _Float16* __restrict__ ctxf, const _Float16* __restrict__ Wot,
                const float* __restrict__ bo, float* __restrict__ out) {
  int lane = threadIdx.x & 31;
  int wf   = blockIdx.x * 4 + (threadIdx.x >> 5);   // 0..2047
  int mt   = wf >> 3;                                // 256 row-tiles of 32
  int nt   = wf & 7;                                 // 8 col-tiles of 32
  int g = lane >> 4, col = lane & 15;

  v8f c[2][2];
  gemm_tile_32x32(ctxf, Wot, mt * 32, nt * 32, lane, c[0][0], c[0][1], c[1][0], c[1][1]);

#pragma unroll
  for (int sn = 0; sn < 2; ++sn) {
    int   n    = nt * 32 + sn * 16 + col;
    float bval = bo[n];
#pragma unroll
    for (int sm = 0; sm < 2; ++sm) {
#pragma unroll
      for (int r = 0; r < 8; ++r) {
        int gm = mt * 32 + sm * 16 + r + 8 * g;
        out[(size_t)gm * 256 + n] = c[sm][sn][r] + bval;
      }
    }
  }
}

// ---------------------------------------------------------------------------
extern "C" void kernel_launch(void* const* d_in, const int* in_sizes, int n_in,
                              void* d_out, int out_size, void* d_ws, size_t ws_size,
                              hipStream_t stream) {
  const float* hid  = (const float*)d_in[0];
  const float* pos  = (const float*)d_in[1];
  const float* mask = (const float*)d_in[2];
  const float* Wq   = (const float*)d_in[3];
  const float* bq   = (const float*)d_in[4];
  const float* Wk   = (const float*)d_in[5];
  const float* bk   = (const float*)d_in[6];
  const float* Wv   = (const float*)d_in[7];
  const float* bv   = (const float*)d_in[8];
  const float* Wo   = (const float*)d_in[9];
  const float* bo   = (const float*)d_in[10];
  float* out = (float*)d_out;

  char*  ws  = (char*)d_ws;
  size_t off = 0;
  auto carve = [&](size_t bytes) {
    void* p = ws + off;
    off = (off + bytes + 255) & ~(size_t)255;
    return p;
  };
  const size_t actB = (size_t)B_ * S_ * D_ * sizeof(_Float16);   // 4 MB
  const size_t wB   = (size_t)256 * 256 * sizeof(_Float16);      // 128 KB
  _Float16* hpf = (_Float16*)carve(actB);
  _Float16* hf  = (_Float16*)carve(actB);
  _Float16* Wqt = (_Float16*)carve(wB);
  _Float16* Wkt = (_Float16*)carve(wB);
  _Float16* Wvt = (_Float16*)carve(wB);
  _Float16* Wot = (_Float16*)carve(wB);
  _Float16* Qf  = (_Float16*)carve(actB);
  _Float16* Kf  = (_Float16*)carve(actB);
  _Float16* Vtv = (_Float16*)carve(actB);
  _Float16* ctxf = hpf;   // hp is consumed by qkv_proj before attention writes ctx

  pack_act_kernel<<<2048, 256, 0, stream>>>(hid, pos, hpf, hf, B_ * S_ * D_);
  pack_w_kernel<<<256, 256, 0, stream>>>(Wq, Wk, Wv, Wo, Wqt, Wkt, Wvt, Wot);
  qkv_proj_kernel<<<1536, 128, 0, stream>>>(hpf, hf, Wqt, Wkt, Wvt,
                                            bq, bk, bv, Qf, Kf, Vtv);
  flash_attn_kernel<<<1024, 128, 0, stream>>>(Qf, Kf, Vtv, mask, ctxf);
  out_proj_kernel<<<512, 128, 0, stream>>>(ctxf, Wot, bo, out);
}
